// DecoderAttentionRNN_13383118094581
// MI455X (gfx1250) — compile-verified, matrix operand order checked by reading, not run on hardware
//
#include <hip/hip_runtime.h>

// ---- problem constants ----
#define M_B     128
#define S_LEN   40
#define T_STEPS 39
#define D_ENC   1280
#define H_DIM   512
#define K_ATT   512
#define E_DIM   512
#define V_SZ    10000
#define L_PIX   196
#define G4H     2048   // 4*H
#define KCAT    2304   // (E + D) + H = 1792 + 512

typedef __bf16 bf16_t;
typedef __attribute__((ext_vector_type(16))) __bf16    v16bf;
typedef __attribute__((ext_vector_type(8)))  float     v8f;
typedef __attribute__((ext_vector_type(4)))  uint32_t  u32x4;
typedef __attribute__((ext_vector_type(8)))  uint32_t  u32x8;

__device__ __forceinline__ float sigmoidf_(float x) { return 1.f / (1.f + __expf(-x)); }

// ---------------------------------------------------------------------------
// Stable descending sort by caption length (M=128, one block) + emit outputs
// ---------------------------------------------------------------------------
__global__ __launch_bounds__(128) void sort_kernel(
    const int* __restrict__ caplen, const int* __restrict__ caps,
    int* __restrict__ sidx, int* __restrict__ dlen,
    float* __restrict__ sidx_out, float* __restrict__ dlen_out,
    float* __restrict__ caps_out)
{
  __shared__ int s_s[M_B];
  __shared__ int s_d[M_B];
  int i = threadIdx.x;
  int li = caplen[i];
  int rank = 0;
  for (int j = 0; j < M_B; ++j) {
    int lj = caplen[j];
    rank += (lj > li) || (lj == li && j < i);
  }
  s_s[rank] = i;
  s_d[rank] = li - 1;
  __syncthreads();
  sidx[i] = s_s[i];
  dlen[i] = s_d[i];
  sidx_out[i] = (float)s_s[i];
  dlen_out[i] = (float)s_d[i];
  for (int idx = i; idx < M_B * S_LEN; idx += M_B) {
    int p = idx / S_LEN, s = idx - p * S_LEN;
    caps_out[idx] = (float)caps[s_s[p] * S_LEN + s];
  }
}

// ---------------------------------------------------------------------------
// fp32 -> bf16 weight conversion (done once; 39 steps then stream bf16)
// ---------------------------------------------------------------------------
__global__ __launch_bounds__(256) void cvt_bf16_kernel(
    const float* __restrict__ src, bf16_t* __restrict__ dst, int n)
{
  for (int i = blockIdx.x * 256 + threadIdx.x; i < n; i += gridDim.x * 256)
    dst[i] = (bf16_t)src[i];
}

// Wg = [W_ih | W_hh] packed along K (2048 x 2304), bf16
__global__ __launch_bounds__(256) void build_wg_kernel(
    const float* __restrict__ W_ih, const float* __restrict__ W_hh,
    bf16_t* __restrict__ Wg)
{
  const int n = G4H * KCAT;
  for (int i = blockIdx.x * 256 + threadIdx.x; i < n; i += gridDim.x * 256) {
    int row = i / KCAT, col = i - row * KCAT;
    float v = (col < 1792) ? W_ih[(size_t)row * 1792 + col]
                           : W_hh[(size_t)row * H_DIM + (col - 1792)];
    Wg[i] = (bf16_t)v;
  }
}

__global__ void combine_bias_kernel(const float* a, const float* b, float* o, int n) {
  int i = blockIdx.x * blockDim.x + threadIdx.x;
  if (i < n) o[i] = a[i] + b[i];
}

// ---------------------------------------------------------------------------
// h0/c0: per-batch mean over 196 pixels, then two 512x1280 GEMVs (fp32 VALU)
// ---------------------------------------------------------------------------
__global__ __launch_bounds__(256) void init_h0c0_kernel(
    const float* __restrict__ enc, const int* __restrict__ sidx,
    const float* __restrict__ W_h0, const float* __restrict__ b_h0,
    const float* __restrict__ W_c0, const float* __restrict__ b_c0,
    float* __restrict__ h, float* __restrict__ c)
{
  __shared__ float mean_s[D_ENC];
  int ms = blockIdx.x;
  int morig = sidx[ms];
  const float* base = enc + (size_t)morig * L_PIX * D_ENC;
  for (int d = threadIdx.x; d < D_ENC; d += 256) {
    float s = 0.f;
    for (int l = 0; l < L_PIX; ++l) s += base[(size_t)l * D_ENC + d];
    mean_s[d] = s * (1.f / L_PIX);
  }
  __syncthreads();
  for (int j = threadIdx.x; j < H_DIM; j += 256) {
    float hh = b_h0[j], cc = b_c0[j];
    const float* wh = W_h0 + (size_t)j * D_ENC;
    const float* wc = W_c0 + (size_t)j * D_ENC;
    for (int k = 0; k < D_ENC; ++k) { float m = mean_s[k]; hh += m * wh[k]; cc += m * wc[k]; }
    h[ms * H_DIM + j] = hh;
    c[ms * H_DIM + j] = cc;
  }
}

// ---------------------------------------------------------------------------
// WMMA bf16 GEMM:  C[M x N] = A[M x K] * W[N x K]^T (+ bias) ; acc fp32
//   block = 256 threads = 8 wave32; blockIdx.y -> 16-row M tile shared via LDS
//   each wave owns one 16x16 output tile -> v_wmma_f32_16x16x32_bf16 chain
// A-panel staging:
//   A_FP32 = true : cooperative fp32->bf16 convert into LDS (enc_a GEMM)
//   A_FP32 = false: Tensor Data Mover (tensor_load_to_lds) DMA of the
//                   16x128 bf16 tile, tracked with TENSORcnt (ISA ch.7/8)
// MODE 0: store fp32 C (ldc = N), bias optional
// MODE 1: fc epilogue -> d_out pred_scores, masked by (t < decode_length[m])
// ---------------------------------------------------------------------------
template <bool A_FP32, int MODE>
__global__ __launch_bounds__(256) void gemm_bf16_wmma(
    const void* __restrict__ Aptr, const bf16_t* __restrict__ W,
    const float* __restrict__ bias, float* __restrict__ Cptr,
    int K, int N, int n_tiles_total,
    const int* __restrict__ d_len, int t)
{
  __shared__ __align__(128) bf16_t As[16 * 128];
  const int tid  = threadIdx.x;
  const int wave = tid >> 5;
  const int lane = tid & 31;
  const int m0   = blockIdx.y * 16;
  const int n_tile = blockIdx.x * 8 + wave;
  const bool valid = n_tile < n_tiles_total;           // wave-uniform: EXEC stays all-ones for WMMA
  const int n0   = n_tile * 16;
  const int brow = n0 + (lane & 15);
  const int bk   = (lane >> 4) * 16;
  const int arow = lane & 15;
  const int ak   = (lane >> 4) * 16;

  v8f acc = {};
  for (int k0 = 0; k0 < K; k0 += 128) {
    if constexpr (A_FP32) {
      // cooperative A panel load -> LDS with on-the-fly fp32 -> bf16 convert
      int e   = tid * 8;
      int row = e >> 7;
      int col = e & 127;
      const float* A = (const float*)Aptr;
      const float* src = A + (size_t)(m0 + row) * K + k0 + col;
#pragma unroll
      for (int i = 0; i < 8; ++i) As[row * 128 + col + i] = (bf16_t)src[i];
    } else {
      // TDM: one tensor_load_to_lds per block per K-chunk (issued by wave 0).
      // D# per CDNA5 ISA:
      //   group0: count=1 | lds_addr | global_addr | type=2
      //   group1: data_size=2B, tensor 128x16 (stride0 = K), tile 128x16
      if (wave == 0) {
        const bf16_t* A = (const bf16_t*)Aptr;
        uint64_t ga  = (uint64_t)(uintptr_t)(A + (size_t)m0 * K + k0);
        uint32_t lds = (uint32_t)(uintptr_t)As;   // generic LDS ptr: low 32 bits = LDS byte addr
        u32x4 g0;
        g0[0] = 1u;                                            // count = 1 valid descriptor
        g0[1] = lds;                                           // lds_addr
        g0[2] = (uint32_t)ga;                                  // global_addr[31:0]
        g0[3] = (uint32_t)((ga >> 32) & 0x01ffffffu)           // global_addr[56:32]
              | 0x80000000u;                                   // type = 2 ("image")
        u32x8 g1;
        g1[0] = (1u << 16);          // wg_mask=0, data_size=1 (2 bytes)
        g1[1] = (128u << 16);        // tensor_dim0 = 128 (lo16 in [31:16])
        g1[2] = (16u << 16);         // tensor_dim0 hi=0, tensor_dim1 = 16 (lo16)
        g1[3] = (128u << 16);        // tensor_dim1 hi=0, tile_dim0 = 128
        g1[4] = 16u;                 // tile_dim1 = 16, tile_dim2 = 0
        g1[5] = (uint32_t)K;         // tensor_dim0_stride (lo32) = K elements
        g1[6] = 0u;                  // stride0 hi, stride1 lo
        g1[7] = 0u;                  // stride1 hi
        asm volatile("tensor_load_to_lds %0, %1" :: "s"(g0), "s"(g1) : "memory");
        __builtin_amdgcn_s_wait_tensorcnt(0);
      }
    }
    __syncthreads();
    if (valid) {
      __builtin_prefetch(W + (size_t)brow * K + k0 + 128, 0, 1);   // global_prefetch next B chunk
#pragma unroll
      for (int kk = 0; kk < 128; kk += 32) {
        v16bf a = *(const v16bf*)&As[arow * 128 + kk + ak];
        v16bf b = *(const v16bf*)(W + (size_t)brow * K + k0 + kk + bk);
        acc = __builtin_amdgcn_wmma_f32_16x16x32_bf16(false, a, false, b,
                                                      (short)0, acc, false, false);
      }
    }
    __syncthreads();
  }

  if (valid) {
    const int col   = lane & 15;
    const int rbase = (lane >> 4) * 8;     // ISA C/D layout: lanes 16-31 hold M = r+8
    float bv = bias ? bias[n0 + col] : 0.f;
#pragma unroll
    for (int r = 0; r < 8; ++r) {
      int row = m0 + rbase + r;
      float v = acc[r] + bv;
      if constexpr (MODE == 0) {
        Cptr[(size_t)row * N + n0 + col] = v;
      } else {
        bool active = (t < d_len[row]);
        Cptr[((size_t)row * T_STEPS + t) * V_SZ + n0 + col] = active ? v : 0.f;
      }
    }
  }
}

// ---------------------------------------------------------------------------
// Fused attention step: dec_a, beta, e=relu(enc_a+dec_a)@W_att, softmax, awe,
// masked alpha write, and builds bf16 xh = [emb_t | beta*awe | h]
// One block per batch row; enc/enc_a indexed through sorted_idx (no gathers).
// ---------------------------------------------------------------------------
__global__ __launch_bounds__(256) void attention_step_kernel(
    const float* __restrict__ enc, const float* __restrict__ enc_a,
    const float* __restrict__ h,
    const float* __restrict__ W_dec, const float* __restrict__ b_dec,
    const float* __restrict__ W_att, const float* __restrict__ b_att,
    const float* __restrict__ W_beta, const float* __restrict__ b_beta,
    const float* __restrict__ emb, const int* __restrict__ caps,
    const int* __restrict__ sidx, const int* __restrict__ dlen,
    bf16_t* __restrict__ xh, float* __restrict__ alpha_out, int t)
{
  __shared__ float h_s[H_DIM];
  __shared__ float dec_s[K_ATT];
  __shared__ float e_s[L_PIX];
  __shared__ float red[256];
  __shared__ float beta_sh;
  const int tid = threadIdx.x;
  const int ms = blockIdx.x;
  const int morig = sidx[ms];

  for (int k = tid; k < H_DIM; k += 256) h_s[k] = h[ms * H_DIM + k];
  __syncthreads();

  // dec_a = h @ W_dec^T + b_dec
  for (int j = tid; j < K_ATT; j += 256) {
    const float* w = W_dec + (size_t)j * H_DIM;
    float s = b_dec[j];
    for (int k = 0; k < H_DIM; ++k) s += h_s[k] * w[k];
    dec_s[j] = s;
  }
  // beta = sigmoid(h . W_beta + b_beta)
  {
    float p = 0.f;
    for (int k = tid; k < H_DIM; k += 256) p += h_s[k] * W_beta[k];
    red[tid] = p;
    __syncthreads();
    for (int s = 128; s > 0; s >>= 1) { if (tid < s) red[tid] += red[tid + s]; __syncthreads(); }
    if (tid == 0) beta_sh = sigmoidf_(red[0] + b_beta[0]);
    __syncthreads();
  }

  // e[l] = relu(enc_a[l] + dec_a) . W_att + b_att
  if (tid < L_PIX) {
    const float* ea = enc_a + ((size_t)morig * L_PIX + tid) * K_ATT;
    float s = b_att[0];
    for (int k = 0; k < K_ATT; ++k) {
      float v = ea[k] + dec_s[k];
      s += fmaxf(v, 0.f) * W_att[k];
    }
    e_s[tid] = s;
  }
  __syncthreads();

  // softmax over 196
  red[tid] = (tid < L_PIX) ? e_s[tid] : -3.4e38f;
  __syncthreads();
  for (int s = 128; s > 0; s >>= 1) { if (tid < s) red[tid] = fmaxf(red[tid], red[tid + s]); __syncthreads(); }
  float mx = red[0];
  __syncthreads();
  float ex = 0.f;
  if (tid < L_PIX) ex = __expf(e_s[tid] - mx);
  red[tid] = ex;
  __syncthreads();
  for (int s = 128; s > 0; s >>= 1) { if (tid < s) red[tid] += red[tid + s]; __syncthreads(); }
  float inv = 1.f / red[0];
  __syncthreads();
  if (tid < L_PIX) e_s[tid] = ex * inv;
  __syncthreads();

  const bool active = (t < dlen[ms]);
  if (tid < L_PIX)
    alpha_out[((size_t)ms * T_STEPS + t) * L_PIX + tid] = active ? e_s[tid] : 0.f;

  // awe[d] = sum_l alpha[l]*enc[l,d]; xh row = [emb_t | beta*awe | h] (bf16)
  const float beta = beta_sh;
  const float* ebase = enc + (size_t)morig * L_PIX * D_ENC;
  bf16_t* xrow = xh + (size_t)ms * KCAT;
  for (int d = tid; d < D_ENC; d += 256) {
    float s = 0.f;
    for (int l = 0; l < L_PIX; ++l) s += e_s[l] * ebase[(size_t)l * D_ENC + d];
    xrow[E_DIM + d] = (bf16_t)(beta * s);
  }
  int cap = caps[morig * S_LEN + t];
  const float* erow = emb + (size_t)cap * E_DIM;
  for (int k = tid; k < E_DIM; k += 256) xrow[k] = (bf16_t)erow[k];
  for (int k = tid; k < H_DIM; k += 256) xrow[1792 + k] = (bf16_t)h_s[k];
}

// ---------------------------------------------------------------------------
// LSTM elementwise: i,f,g,o split of gates; update h/c; emit bf16 h for fc
// ---------------------------------------------------------------------------
__global__ __launch_bounds__(256) void lstm_step_kernel(
    const float* __restrict__ gates,
    float* __restrict__ h, float* __restrict__ c, bf16_t* __restrict__ h_bf)
{
  int m = blockIdx.x;
  const float* g = gates + (size_t)m * G4H;
  for (int j = threadIdx.x; j < H_DIM; j += 256) {
    float ig = sigmoidf_(g[j]);
    float fg = sigmoidf_(g[j + 512]);
    float gg = tanhf(g[j + 1024]);
    float og = sigmoidf_(g[j + 1536]);
    float cn = fg * c[m * H_DIM + j] + ig * gg;
    float hn = og * tanhf(cn);
    c[m * H_DIM + j] = cn;
    h[m * H_DIM + j] = hn;
    h_bf[m * H_DIM + j] = (bf16_t)hn;
  }
}

// ---------------------------------------------------------------------------
extern "C" void kernel_launch(void* const* d_in, const int* in_sizes, int n_in,
                              void* d_out, int out_size, void* d_ws, size_t ws_size,
                              hipStream_t stream)
{
  (void)in_sizes; (void)n_in; (void)out_size; (void)ws_size;
  const float* enc    = (const float*)d_in[0];
  const int*   caps   = (const int*)d_in[1];
  const int*   caplen = (const int*)d_in[2];
  const float* W_enc  = (const float*)d_in[3];
  const float* b_enc  = (const float*)d_in[4];
  const float* W_dec  = (const float*)d_in[5];
  const float* b_dec  = (const float*)d_in[6];
  const float* W_att  = (const float*)d_in[7];
  const float* b_att  = (const float*)d_in[8];
  const float* emb    = (const float*)d_in[9];
  const float* W_ih   = (const float*)d_in[10];
  const float* b_ih   = (const float*)d_in[11];
  const float* W_hh   = (const float*)d_in[12];
  const float* b_hh   = (const float*)d_in[13];
  const float* W_h0   = (const float*)d_in[14];
  const float* b_h0   = (const float*)d_in[15];
  const float* W_c0   = (const float*)d_in[16];
  const float* b_c0   = (const float*)d_in[17];
  const float* W_beta = (const float*)d_in[18];
  const float* b_beta = (const float*)d_in[19];
  const float* W_fc   = (const float*)d_in[20];
  const float* b_fc   = (const float*)d_in[21];

  // output regions (flat float, in reference return order)
  float* out       = (float*)d_out;
  float* pred_out  = out;                                   // 128*39*10000
  float* caps_out  = out + 49920000;                        // 128*40
  float* dlen_out  = out + 49925120;                        // 128
  float* alpha_out = out + 49925248;                        // 128*39*196
  float* sidx_out  = out + 50903680;                        // 128

  // workspace carve-out (~75 MB)
  char* ws = (char*)d_ws;
  size_t off = 0;
  auto take = [&](size_t bytes) -> char* {
    char* p = ws + off;
    off = (off + bytes + 255) & ~(size_t)255;
    return p;
  };
  int*    sidx_i   = (int*)take(M_B * 4);
  int*    dlen_i   = (int*)take(M_B * 4);
  float*  h_ws     = (float*)take((size_t)M_B * H_DIM * 4);
  float*  c_ws     = (float*)take((size_t)M_B * H_DIM * 4);
  float*  gates_ws = (float*)take((size_t)M_B * G4H * 4);
  bf16_t* hbf_ws   = (bf16_t*)take((size_t)M_B * H_DIM * 2);
  bf16_t* xh_ws    = (bf16_t*)take((size_t)M_B * KCAT * 2);
  float*  bg_ws    = (float*)take((size_t)G4H * 4);
  bf16_t* wenc_bf  = (bf16_t*)take((size_t)K_ATT * D_ENC * 2);
  bf16_t* wg_bf    = (bf16_t*)take((size_t)G4H * KCAT * 2);
  bf16_t* wfc_bf   = (bf16_t*)take((size_t)V_SZ * H_DIM * 2);
  float*  enca_ws  = (float*)take((size_t)M_B * L_PIX * K_ATT * 4);

  // 1) sort + emit sorted_idx / decode_lengths / sorted caps
  sort_kernel<<<1, 128, 0, stream>>>(caplen, caps, sidx_i, dlen_i,
                                     sidx_out, dlen_out, caps_out);

  // 2) weight prep (bf16 conversion / packing) — once per launch
  cvt_bf16_kernel<<<1024, 256, 0, stream>>>(W_enc, wenc_bf, K_ATT * D_ENC);
  cvt_bf16_kernel<<<2048, 256, 0, stream>>>(W_fc,  wfc_bf, V_SZ * H_DIM);
  build_wg_kernel<<<2048, 256, 0, stream>>>(W_ih, W_hh, wg_bf);
  combine_bias_kernel<<<8, 256, 0, stream>>>(b_ih, b_hh, bg_ws, G4H);

  // 3) h0/c0
  init_h0c0_kernel<<<128, 256, 0, stream>>>(enc, sidx_i, W_h0, b_h0, W_c0, b_c0,
                                            h_ws, c_ws);

  // 4) enc_a = enc @ W_enc^T + b_enc  (25088 x 1280 x 512), raw order
  gemm_bf16_wmma<true, 0><<<dim3(4, 1568), 256, 0, stream>>>(
      enc, wenc_bf, b_enc, enca_ws, D_ENC, K_ATT, 32, nullptr, 0);

  // 5) sequential decode: 39 steps x {attention, gates GEMM, LSTM, fc GEMM}
  for (int t = 0; t < T_STEPS; ++t) {
    attention_step_kernel<<<128, 256, 0, stream>>>(
        enc, enca_ws, h_ws, W_dec, b_dec, W_att, b_att, W_beta, b_beta,
        emb, caps, sidx_i, dlen_i, xh_ws, alpha_out, t);

    // gates = [x_in | h] @ [W_ih | W_hh]^T + (b_ih + b_hh)   (128 x 2304 x 2048)
    gemm_bf16_wmma<false, 0><<<dim3(16, 8), 256, 0, stream>>>(
        xh_ws, wg_bf, bg_ws, gates_ws, KCAT, G4H, 128, nullptr, 0);

    lstm_step_kernel<<<128, 256, 0, stream>>>(gates_ws, h_ws, c_ws, hbf_ws);

    // scores = h_new @ W_fc^T + b_fc, masked -> d_out  (128 x 512 x 10000)
    gemm_bf16_wmma<false, 1><<<dim3(79, 8), 256, 0, stream>>>(
        hbf_ws, wfc_bf, b_fc, pred_out, H_DIM, V_SZ, 625, dlen_i, t);
  }
}
